// TokenPose_26671746908305
// MI455X (gfx1250) — matrix operand order repaired and checked by reading, not compiled, not run on hardware
//
#include <hip/hip_runtime.h>

typedef __attribute__((ext_vector_type(16))) _Float16 v16h;
typedef __attribute__((ext_vector_type(4)))  _Float16 v4h;
typedef __attribute__((ext_vector_type(8)))  float    v8f;

#define NTOK 1728
#define DMODEL 384
#define FDIM 1536
#define NHEAD 12
#define GRID3 12
#define SVOX 48

static __device__ inline v8f wmma32(v16h a, v16h b, v8f c) {
  // D = A(16x32 f16) * B(32x16 f16) + C(16x16 f32)
  return __builtin_amdgcn_wmma_f32_16x16x32_f16(false, a, false, b, (short)0, c,
                                                false, false);
}

// ---------------------------------------------------------------------------
// Repack wq/wk/wv [L,H,D,DH] -> [L, D, H*DH] so QKV is one [384x384] GEMM.
// ---------------------------------------------------------------------------
__global__ __launch_bounds__(256) void repack_kernel(
    const float* __restrict__ wq, const float* __restrict__ wk,
    const float* __restrict__ wv, float* __restrict__ oq,
    float* __restrict__ ok_, float* __restrict__ ov) {
  int idx = blockIdx.x * 256 + threadIdx.x;
  const int TOT = 8 * NHEAD * DMODEL * 32;
  if (idx >= TOT) return;
  int k  = idx & 31;
  int d  = (idx >> 5) % DMODEL;
  int hh = ((idx >> 5) / DMODEL) % NHEAD;
  int l  = (idx >> 5) / (DMODEL * NHEAD);
  int dst = (l * DMODEL + d) * DMODEL + hh * 32 + k;
  oq[dst] = wq[idx];
  ok_[dst] = wk[idx];
  ov[dst] = wv[idx];
}

// ---------------------------------------------------------------------------
// Patch embedding (Conv3d 1->384, k=4,s=4) + sinusoidal posenc -> t[N, D]
// ---------------------------------------------------------------------------
__global__ __launch_bounds__(128) void patch_kernel(
    const float* __restrict__ x, const float* __restrict__ pw,
    const float* __restrict__ pb, float* __restrict__ t) {
  __shared__ float xs[64];
  int n = blockIdx.x;
  int gx = n / (GRID3 * GRID3), gy = (n / GRID3) % GRID3, gz = n % GRID3;
  int tid = threadIdx.x;
  if (tid < 64) {
    int p = tid >> 4, q = (tid >> 2) & 3, r = tid & 3;
    xs[tid] = x[(long)(gx * 4 + p) * SVOX * SVOX + (gy * 4 + q) * SVOX +
                (gz * 4 + r)];
  }
  __syncthreads();
  for (int c = tid; c < DMODEL; c += 128) {
    float acc = pb[c];
    const float* w = pw + c * 64;
#pragma unroll 8
    for (int i = 0; i < 64; ++i) acc += xs[i] * w[i];
    int a = c >> 7, cc = c & 127, ii = cc >> 1;
    float pos = (a == 0) ? (float)gx : (a == 1) ? (float)gy : (float)gz;
    float freq = __powf(10000.f, -(float)(2 * ii) / 128.f);
    float ang = pos * freq;
    acc += (cc & 1) ? __cosf(ang) : __sinf(ang);
    t[(long)n * DMODEL + c] = acc;
  }
}

// ---------------------------------------------------------------------------
// LayerNorm over last dim (D=384)
// ---------------------------------------------------------------------------
__global__ __launch_bounds__(128) void ln_kernel(
    const float* __restrict__ X, const float* __restrict__ g,
    const float* __restrict__ b, float* __restrict__ Y, int D) {
  __shared__ float red[128];
  int row = blockIdx.x, tid = threadIdx.x;
  const float* xr = X + (long)row * D;
  float s = 0.f;
  for (int c = tid; c < D; c += 128) s += xr[c];
  red[tid] = s;
  __syncthreads();
  for (int st = 64; st > 0; st >>= 1) {
    if (tid < st) red[tid] += red[tid + st];
    __syncthreads();
  }
  float mean = red[0] / D;
  __syncthreads();
  float vs = 0.f;
  for (int c = tid; c < D; c += 128) {
    float d = xr[c] - mean;
    vs += d * d;
  }
  red[tid] = vs;
  __syncthreads();
  for (int st = 64; st > 0; st >>= 1) {
    if (tid < st) red[tid] += red[tid + st];
    __syncthreads();
  }
  float rstd = rsqrtf(red[0] / D + 1e-5f);
  for (int c = tid; c < D; c += 128)
    Y[(long)row * D + c] = (xr[c] - mean) * rstd * g[c] + b[c];
}

// ---------------------------------------------------------------------------
// Tiled WMMA GEMM: Out[M,N] = act(A[M,K] * W[K,N] + bias) (+resid), optional
// deconv scatter of output rows. fp32 in/out, f16 WMMA compute.
// Block tile 128x64, K-step 32, 8 waves (4x2), each wave 2x2 WMMA tiles.
// Software-pipelined: next K-tile global loads are issued between the two
// barriers so they overlap the ds_load/v_wmma work of the current tile.
// DENSE_B is a compile-time switch (dense colStride==1 vs strided deconv
// weights) so the staging block is straight-line code per instantiation.
// ---------------------------------------------------------------------------
template <bool DENSE_B>
__global__ __launch_bounds__(256) void gemm_kernel(
    const float* __restrict__ A, int lda, const float* __restrict__ Bw,
    long ldb, int colStride, const float* __restrict__ bias,
    const float* __restrict__ resid, float* __restrict__ Out, int M, int Nn,
    int Kk, int flags, int scatterG, int offp, int offq, int offr) {
  __shared__ _Float16 As[128][32];   // [row][k]   row stride 64B (16B aligned)
  __shared__ _Float16 Bs[64][40];    // [n][k]     row stride 80B (16B aligned)
  int tid = threadIdx.x;
  int wave = tid >> 5, lane = tid & 31;
  int wm = wave & 3, wn = wave >> 2;
  int blockM = blockIdx.y * 128, blockN = blockIdx.x * 64;
  int hf = lane >> 4, l16 = lane & 15;

  v8f acc[2][2] = {};

  float4 fa[4];   // staged A regs (4 x float4 per thread)
  float4 fb[2];   // staged B regs (dense path)
  float  fbs[8];  // staged B regs (strided path)

  auto loadA = [&](int kt) {
#pragma unroll
    for (int i = 0; i < 4; ++i) {
      int chunk = i * 256 + tid;  // 0..1023
      int row = chunk >> 3;
      int kc = (chunk & 7) * 4;
      int rowg = blockM + row;
      rowg = rowg < M ? rowg : M - 1;
      fa[i] = *(const float4*)(A + (long)rowg * lda + kt + kc);
    }
  };
  auto loadB = [&](int kt) {
    if (DENSE_B) {  // dense weights, Nn % 64 == 0
#pragma unroll
      for (int i = 0; i < 2; ++i) {
        int chunk = i * 256 + tid;  // 0..511
        int kk = chunk >> 4;        // 0..31
        int nc = (chunk & 15) * 4;  // 0..60
        fb[i] = *(const float4*)(Bw + (long)(kt + kk) * ldb + blockN + nc);
      }
    } else {  // strided weights (deconv), clamped cols
#pragma unroll
      for (int i = 0; i < 8; ++i) {
        int idx = i * 256 + tid;
        int kk = idx >> 6, n = idx & 63;
        int ng = blockN + n;
        ng = ng < Nn ? ng : Nn - 1;
        fbs[i] = Bw[(long)(kt + kk) * ldb + (long)ng * colStride];
      }
    }
  };

  const int kLast = Kk - 32;
  loadA(0);
  loadB(0);

  for (int k0 = 0; k0 < Kk; k0 += 32) {
    // commit staged regs to LDS (f32 -> f16)
#pragma unroll
    for (int i = 0; i < 4; ++i) {
      int chunk = i * 256 + tid;
      int row = chunk >> 3;
      int kc = (chunk & 7) * 4;
      v4h h4 = {(_Float16)fa[i].x, (_Float16)fa[i].y, (_Float16)fa[i].z,
                (_Float16)fa[i].w};
      *(v4h*)&As[row][kc] = h4;
    }
    if (DENSE_B) {
#pragma unroll
      for (int i = 0; i < 2; ++i) {
        int chunk = i * 256 + tid;
        int kk = chunk >> 4;
        int nc = (chunk & 15) * 4;
        Bs[nc + 0][kk] = (_Float16)fb[i].x;
        Bs[nc + 1][kk] = (_Float16)fb[i].y;
        Bs[nc + 2][kk] = (_Float16)fb[i].z;
        Bs[nc + 3][kk] = (_Float16)fb[i].w;
      }
    } else {
#pragma unroll
      for (int i = 0; i < 8; ++i) {
        int idx = i * 256 + tid;
        int kk = idx >> 6, n = idx & 63;
        Bs[n][kk] = (_Float16)fbs[i];
      }
    }
    __syncthreads();

    // issue next tile's global loads (clamped; overlap with WMMA below)
    int kn = (k0 + 32 <= kLast) ? (k0 + 32) : kLast;
    __builtin_prefetch(A + (long)((blockM + (tid >> 1)) < M
                                      ? (blockM + (tid >> 1))
                                      : (M - 1)) * lda +
                           kn + ((tid & 1) << 4),
                       0, 0);
    loadA(kn);
    loadB(kn);

    v16h a0, a1, b0, b1;
#pragma unroll
    for (int e = 0; e < 16; ++e) {
      int k = (hf ? 16 : 0) + e;                         // A-frag K split
      a0[e] = As[wm * 32 + l16][k];
      a1[e] = As[wm * 32 + 16 + l16][k];
      int kb = (hf ? 16 : 0) + e;                        // B-frag K split
      b0[e] = Bs[wn * 32 + l16][kb];
      b1[e] = Bs[wn * 32 + 16 + l16][kb];
    }
    acc[0][0] = wmma32(a0, b0, acc[0][0]);
    acc[0][1] = wmma32(a0, b1, acc[0][1]);
    acc[1][0] = wmma32(a1, b0, acc[1][0]);
    acc[1][1] = wmma32(a1, b1, acc[1][1]);
    __syncthreads();
  }

#pragma unroll
  for (int j = 0; j < 2; ++j) {
    int gcol = blockN + wn * 32 + j * 16 + l16;
    if (gcol >= Nn) continue;
    float bv = (flags & 1) ? bias[gcol] : 0.f;
#pragma unroll
    for (int i = 0; i < 2; ++i) {
#pragma unroll
      for (int v = 0; v < 8; ++v) {
        int grow = blockM + wm * 32 + i * 16 + (hf ? v + 8 : v);
        if (grow >= M) continue;
        float val = acc[i][j][v] + bv;
        if (flags & 2) val = fmaxf(val, 0.f);
        if (flags & 4) val += resid[(long)grow * Nn + gcol];
        long orow = grow;
        if (scatterG > 0) {
          int G = scatterG, G2 = 2 * G;
          int xx = grow / (G * G);
          int rem = grow - xx * G * G;
          int yy = rem / G, zz = rem - yy * G;
          orow = ((long)(2 * xx + offp) * G2 + (2 * yy + offq)) * G2 +
                 (2 * zz + offr);
        }
        Out[orow * Nn + gcol] = val;
      }
    }
  }
}

// ---------------------------------------------------------------------------
// Flash attention: one wave = 16 Q rows of one head, streams 32 keys/iter.
// K/V tiles staged cooperatively by all 4 waves (b128 loads, software-
// pipelined into regs one tile ahead), QK^T and PV via
// v_wmma_f32_16x16x32_f16, online softmax with 16-lane shuffle reductions,
// P re-laid-out C-format -> A-format through LDS.
// ---------------------------------------------------------------------------
__global__ __launch_bounds__(128) void attn_kernel(
    const float* __restrict__ Q, const float* __restrict__ K,
    const float* __restrict__ V, float* __restrict__ O, int N, int ld,
    float scale) {
  __shared__ _Float16 Ks[32][40];       // [key][dh]
  __shared__ _Float16 Vs[32][40];       // [dh][key] (transposed)
  __shared__ _Float16 Plds[4][16][40];  // per-wave P tile
  int tid = threadIdx.x;
  int wave = tid >> 5;
  int lane = tid & 31;
  int h = blockIdx.y;
  int r0 = blockIdx.x * 64 + wave * 16;
  int hf = lane >> 4, l16 = lane & 15;
  int cbase = h * 32;

  v16h qa;
  int qrow = r0 + l16;
#pragma unroll
  for (int e = 0; e < 16; ++e) {
    int k = (hf ? 16 : 0) + e;
    qa[e] = (_Float16)(Q[(long)qrow * ld + cbase + k] * scale);
  }

  v8f oA = {}, oB = {};
  float mrow[8], lrow[8];
#pragma unroll
  for (int v = 0; v < 8; ++v) {
    mrow[v] = -1e30f;
    lrow[v] = 0.f;
  }

  float4 kreg[2], vreg[2];
  auto loadKV = [&](int jt) {
#pragma unroll
    for (int i = 0; i < 2; ++i) {
      int chunk = i * 128 + tid;  // 0..255
      int key = chunk >> 3;       // 0..31
      int dhc = (chunk & 7) * 4;  // 0..28
      kreg[i] = *(const float4*)(K + (long)(jt + key) * ld + cbase + dhc);
      vreg[i] = *(const float4*)(V + (long)(jt + key) * ld + cbase + dhc);
    }
  };
  loadKV(0);

  for (int j0 = 0; j0 < N; j0 += 32) {
    // commit staged K/V regs to LDS (tile shared by all 4 waves)
#pragma unroll
    for (int i = 0; i < 2; ++i) {
      int chunk = i * 128 + tid;
      int key = chunk >> 3;
      int dhc = (chunk & 7) * 4;
      v4h kh = {(_Float16)kreg[i].x, (_Float16)kreg[i].y, (_Float16)kreg[i].z,
                (_Float16)kreg[i].w};
      *(v4h*)&Ks[key][dhc] = kh;
      Vs[dhc + 0][key] = (_Float16)vreg[i].x;
      Vs[dhc + 1][key] = (_Float16)vreg[i].y;
      Vs[dhc + 2][key] = (_Float16)vreg[i].z;
      Vs[dhc + 3][key] = (_Float16)vreg[i].w;
    }
    __syncthreads();

    // issue next tile's loads; they overlap softmax + WMMA below
    int jn = (j0 + 32 < N) ? (j0 + 32) : j0;
    loadKV(jn);

    // K^T fragments: B[k=dh, n=key]
    v16h kfA, kfB;
#pragma unroll
    for (int e = 0; e < 16; ++e) {
      int dh = (hf ? 16 : 0) + e;
      kfA[e] = Ks[l16][dh];
      kfB[e] = Ks[16 + l16][dh];
    }
    v8f zc = {};
    v8f sA = wmma32(qa, kfA, zc);
    v8f sB = wmma32(qa, kfB, zc);

#pragma unroll
    for (int v = 0; v < 8; ++v) {
      float xa = sA[v], xb = sB[v];
      float mx = fmaxf(xa, xb);
#pragma unroll
      for (int off = 8; off >= 1; off >>= 1)
        mx = fmaxf(mx, __shfl_xor(mx, off, 32));
      float mn = fmaxf(mrow[v], mx);
      float corr = __expf(mrow[v] - mn);
      float pa = __expf(xa - mn), pb = __expf(xb - mn);
      float rs = pa + pb;
#pragma unroll
      for (int off = 8; off >= 1; off >>= 1) rs += __shfl_xor(rs, off, 32);
      lrow[v] = lrow[v] * corr + rs;
      mrow[v] = mn;
      oA[v] *= corr;
      oB[v] *= corr;
      int prow = hf ? v + 8 : v;
      Plds[wave][prow][l16] = (_Float16)pa;
      Plds[wave][prow][16 + l16] = (_Float16)pb;
    }
    __syncthreads();  // uniform across all waves (same trip count)

    v16h pf;
#pragma unroll
    for (int e = 0; e < 16; ++e) {
      int k = (hf ? 16 : 0) + e;
      pf[e] = Plds[wave][l16][k];
    }
    v16h vfA, vfB;
#pragma unroll
    for (int e = 0; e < 16; ++e) {
      int kb2 = (hf ? 16 : 0) + e;
      vfA[e] = Vs[l16][kb2];
      vfB[e] = Vs[16 + l16][kb2];
    }
    oA = wmma32(pf, vfA, oA);
    oB = wmma32(pf, vfB, oB);
    __syncthreads();
  }

#pragma unroll
  for (int v = 0; v < 8; ++v) {
    int row = r0 + (hf ? v + 8 : v);
    float inv = 1.f / lrow[v];
    O[(long)row * ld + cbase + l16] = oA[v] * inv;
    O[(long)row * ld + cbase + 16 + l16] = oB[v] * inv;
  }
}

// ---------------------------------------------------------------------------
// Training-mode BatchNorm3d (batch stats over voxels) + ReLU, in place.
// One block per channel.
// ---------------------------------------------------------------------------
__global__ __launch_bounds__(256) void bn_relu_kernel(
    float* __restrict__ Z, const float* __restrict__ g,
    const float* __restrict__ b, int C, int V) {
  __shared__ float s1[256], s2[256];
  int c = blockIdx.x, tid = threadIdx.x;
  float sum = 0.f, sq = 0.f;
  for (int v = tid; v < V; v += 256) {
    float z = Z[(long)v * C + c];
    sum += z;
    sq += z * z;
  }
  s1[tid] = sum;
  s2[tid] = sq;
  __syncthreads();
  for (int st = 128; st > 0; st >>= 1) {
    if (tid < st) {
      s1[tid] += s1[tid + st];
      s2[tid] += s2[tid + st];
    }
    __syncthreads();
  }
  float mean = s1[0] / V;
  float var = s2[0] / V - mean * mean;
  float rstd = rsqrtf(var + 1e-5f);
  float gc = g[c], bc = b[c];
  for (int v = tid; v < V; v += 256) {
    long idx = (long)v * C + c;
    float z = (Z[idx] - mean) * rstd * gc + bc;
    Z[idx] = fmaxf(z, 0.f);
  }
}

// ---------------------------------------------------------------------------
// Final 1x1x1 conv: out[o, vox] = dot(Z[vox,:], out_w[o,:]) + out_b[o]
// ---------------------------------------------------------------------------
__global__ __launch_bounds__(256) void outproj_kernel(
    const float* __restrict__ Z, const float* __restrict__ W,
    const float* __restrict__ bias, float* __restrict__ out, int V, int C) {
  __shared__ float Ws[15 * DMODEL];
  int tid = threadIdx.x;
  for (int i = tid; i < 15 * C; i += 256) Ws[i] = W[i];
  __syncthreads();
  int vox = blockIdx.x * 256 + tid;
  if (vox >= V) return;
  float acc[15];
#pragma unroll
  for (int o = 0; o < 15; ++o) acc[o] = bias[o];
  for (int d = 0; d < C; d += 4) {
    float4 z = *(const float4*)(Z + (long)vox * C + d);
#pragma unroll
    for (int o = 0; o < 15; ++o) {
      const float* wr = &Ws[o * C + d];
      acc[o] += z.x * wr[0] + z.y * wr[1] + z.z * wr[2] + z.w * wr[3];
    }
  }
#pragma unroll
  for (int o = 0; o < 15; ++o) out[(long)o * V + vox] = acc[o];
}

// ---------------------------------------------------------------------------
extern "C" void kernel_launch(void* const* d_in, const int* in_sizes, int n_in,
                              void* d_out, int out_size, void* d_ws,
                              size_t ws_size, hipStream_t stream) {
  const float* x       = (const float*)d_in[0];
  const float* patch_w = (const float*)d_in[1];
  const float* patch_b = (const float*)d_in[2];
  const float* ln1_g   = (const float*)d_in[3];
  const float* ln1_b   = (const float*)d_in[4];
  const float* wq      = (const float*)d_in[5];
  const float* wk      = (const float*)d_in[6];
  const float* wv      = (const float*)d_in[7];
  const float* wo      = (const float*)d_in[8];
  const float* wo_b    = (const float*)d_in[9];
  const float* ln2_g   = (const float*)d_in[10];
  const float* ln2_b   = (const float*)d_in[11];
  const float* w1      = (const float*)d_in[12];
  const float* b1      = (const float*)d_in[13];
  const float* w2      = (const float*)d_in[14];
  const float* b2      = (const float*)d_in[15];
  const float* dec1_w  = (const float*)d_in[16];
  const float* dec1_b  = (const float*)d_in[17];
  const float* bn1_g   = (const float*)d_in[18];
  const float* bn1_b   = (const float*)d_in[19];
  const float* dec2_w  = (const float*)d_in[20];
  const float* dec2_b  = (const float*)d_in[21];
  const float* bn2_g   = (const float*)d_in[22];
  const float* bn2_b   = (const float*)d_in[23];
  const float* out_w   = (const float*)d_in[24];
  const float* out_b   = (const float*)d_in[25];

  float* ws = (float*)d_ws;
  const size_t SZ = (size_t)NTOK * DMODEL;  // 663552
  float* tA  = ws;
  float* hbf = ws + 1 * SZ;
  float* qb  = ws + 2 * SZ;
  float* kb  = ws + 3 * SZ;
  float* vb  = ws + 4 * SZ;
  float* ao  = ws + 5 * SZ;
  float* tB  = ws + 6 * SZ;
  float* mid = ws + 7 * SZ;                       // NTOK*FDIM
  float* wqp = mid + (size_t)NTOK * FDIM;         // 8*384*384 each
  float* wkp = wqp + (size_t)8 * DMODEL * DMODEL;
  float* wvp = wkp + (size_t)8 * DMODEL * DMODEL;
  float* Z2  = wvp + (size_t)8 * DMODEL * DMODEL; // 24^3 * 384
  float* Z3  = Z2 + (size_t)13824 * DMODEL;       // 48^3 * 384

  auto gemm = [&](const float* A, int lda, const float* Bw, long ldb,
                  const float* bias, const float* resid, float* Outp, int M,
                  int Nn, int Kk, int flags) {
    dim3 g((Nn + 63) / 64, (M + 127) / 128);
    gemm_kernel<true><<<g, dim3(256), 0, stream>>>(
        A, lda, Bw, ldb, 1, bias, resid, Outp, M, Nn, Kk, flags, 0, 0, 0, 0);
  };
  auto gemm_scatter = [&](const float* A, const float* Bw, const float* bias,
                          float* Outp, int M, int sG, int op, int oq,
                          int orr) {
    dim3 g((DMODEL + 63) / 64, (M + 127) / 128);
    gemm_kernel<false><<<g, dim3(256), 0, stream>>>(
        A, DMODEL, Bw, (long)DMODEL * 8, 8, bias, nullptr, Outp, M, DMODEL,
        DMODEL, 1, sG, op, oq, orr);
  };

  // 0) repack QKV weights to [L, D, H*DH]
  repack_kernel<<<(8 * NHEAD * DMODEL * 32 + 255) / 256, 256, 0, stream>>>(
      wq, wk, wv, wqp, wkp, wvp);

  // 1) patch embed + posenc -> t
  patch_kernel<<<NTOK, 128, 0, stream>>>(x, patch_w, patch_b, tA);

  // 2) transformer blocks
  const float scale = 0.17677669529663687f;  // 1/sqrt(32)
  for (int l = 0; l < 8; ++l) {
    ln_kernel<<<NTOK, 128, 0, stream>>>(tA, ln1_g + l * DMODEL,
                                        ln1_b + l * DMODEL, hbf, DMODEL);
    const size_t wsz = (size_t)DMODEL * DMODEL;
    gemm(hbf, DMODEL, wqp + l * wsz, DMODEL, nullptr, nullptr, qb, NTOK,
         DMODEL, DMODEL, 0);
    gemm(hbf, DMODEL, wkp + l * wsz, DMODEL, nullptr, nullptr, kb, NTOK,
         DMODEL, DMODEL, 0);
    gemm(hbf, DMODEL, wvp + l * wsz, DMODEL, nullptr, nullptr, vb, NTOK,
         DMODEL, DMODEL, 0);
    attn_kernel<<<dim3(NTOK / 64, NHEAD), 128, 0, stream>>>(qb, kb, vb, ao,
                                                            NTOK, DMODEL,
                                                            scale);
    gemm(ao, DMODEL, wo + l * wsz, DMODEL, wo_b + l * DMODEL, tA, tB, NTOK,
         DMODEL, DMODEL, 1 | 4);
    ln_kernel<<<NTOK, 128, 0, stream>>>(tB, ln2_g + l * DMODEL,
                                        ln2_b + l * DMODEL, hbf, DMODEL);
    gemm(hbf, DMODEL, w1 + (size_t)l * DMODEL * FDIM, FDIM, b1 + l * FDIM,
         nullptr, mid, NTOK, FDIM, DMODEL, 1 | 2);
    gemm(mid, FDIM, w2 + (size_t)l * FDIM * DMODEL, DMODEL, b2 + l * DMODEL,
         tB, tA, NTOK, DMODEL, FDIM, 1 | 4);
  }

  // 3) decoder: ConvTranspose3d(k=2,s=2) = 8 scatter-GEMMs each
  for (int p = 0; p < 2; ++p)
    for (int q = 0; q < 2; ++q)
      for (int r = 0; r < 2; ++r) {
        int off = p * 4 + q * 2 + r;
        gemm_scatter(tA, dec1_w + off, dec1_b, Z2, NTOK, GRID3, p, q, r);
      }
  bn_relu_kernel<<<DMODEL, 256, 0, stream>>>(Z2, bn1_g, bn1_b, DMODEL, 13824);

  for (int p = 0; p < 2; ++p)
    for (int q = 0; q < 2; ++q)
      for (int r = 0; r < 2; ++r) {
        int off = p * 4 + q * 2 + r;
        gemm_scatter(Z2, dec2_w + off, dec2_b, Z3, 13824, 24, p, q, r);
      }
  bn_relu_kernel<<<DMODEL, 256, 0, stream>>>(Z3, bn2_g, bn2_b, DMODEL, 110592);

  // 4) final 1x1 conv -> d_out [1,15,48,48,48]
  outproj_kernel<<<(110592 + 255) / 256, 256, 0, stream>>>(
      Z3, out_w, out_b, (float*)d_out, 110592, DMODEL);
}
